// MLA_30150670418518
// MI455X (gfx1250) — matrix operand order repaired and checked by reading, not compiled, not run on hardware
//
#include <hip/hip_runtime.h>

// ---------------------------------------------------------------------------
// MLA (DeepSeek-style) for MI455X / gfx1250. Compute-bound (~270 GFLOP vs
// ~110MB of L2-resident data @23.3TB/s) => bf16 WMMA (v_wmma_f32_16x16x32_bf16)
// with f32 accumulation. wave32, 8 waves / 256-thread block. GEMMs use
// double-buffered GLOBAL_LOAD_ASYNC_TO_LDS_B128 staging (ASYNCcnt pipeline).
// ---------------------------------------------------------------------------

typedef __bf16 bf16_t;
typedef __attribute__((ext_vector_type(16))) __bf16 v16bf;
typedef __attribute__((ext_vector_type(8)))  __bf16 bf16x8;
typedef __attribute__((ext_vector_type(8)))  float  v8f;

#define DIM 2048
#define H_  16
#define QR_ 1536
#define KVR_ 512
#define DN_ 128
#define DR_ 64
#define DV_ 128
#define BS_ 4
#define S_  1024
#define QKH 192            // DN + DR
#define KCAT 576           // KVR + DR
#define TOK (BS_*S_)       // 4096

__device__ __forceinline__ v16bf make16(bf16x8 lo, bf16x8 hi) {
    return __builtin_shufflevector(lo, hi, 0,1,2,3,4,5,6,7,8,9,10,11,12,13,14,15);
}

__device__ __forceinline__ v8f wmma_bf16(v16bf a, v16bf b, v8f c) {
    return __builtin_amdgcn_wmma_f32_16x16x32_bf16(false, a, false, b, (short)0, c, false, false);
}

// A-fragment (16x32, M x K): lane L holds row m=(L&15); low lanes K {k0..k0+7,
// k0+16..k0+23}, high lanes K {k0+8..k0+15, k0+24..k0+31}  (ISA 7.12.2).
__device__ __forceinline__ v16bf load_a_frag(const bf16_t* rowp, int k) {
    return make16(*(const bf16x8*)(rowp + k), *(const bf16x8*)(rowp + k + 16));
}
// B-fragment (32x16, K x N): lane L holds col n=(L&15); low lanes K {k0..k0+7,
// k0+16..k0+23}, high lanes K {k0+8..k0+15, k0+24..k0+31} (same as A).
__device__ __forceinline__ v16bf load_b_frag(const bf16_t* rowp, int k) {
    return make16(*(const bf16x8*)(rowp + k), *(const bf16x8*)(rowp + k + 16));
}

// Async DMA: global (saddr base + 32-bit vgpr byte offset) -> LDS, 16B/lane.
// Tracked by ASYNCcnt (ISA 10.4 / 15.18.3 op 98).
__device__ __forceinline__ void async_g2l_16B(const void* sbase, unsigned gbyteoff,
                                              unsigned ldsbyteoff) {
    asm volatile("global_load_async_to_lds_b128 %0, %1, %2 offset:0"
                 :: "v"(ldsbyteoff), "v"(gbyteoff), "s"(sbase) : "memory");
}
__device__ __forceinline__ void wait_async_le4() {
    asm volatile("s_wait_asynccnt 0x4" ::: "memory");
}
__device__ __forceinline__ void wait_async_le0() {
    asm volatile("s_wait_asynccnt 0x0" ::: "memory");
}
__device__ __forceinline__ unsigned lds_off32(const void* p) {
    // flat shared address: LDS offset is the low 32 bits (ISA 10.2 aperture rules)
    return (unsigned)(size_t)p;
}

// ---------------------------------------------------------------------------
// f32 -> bf16 convert
// ---------------------------------------------------------------------------
__global__ void cvt_kernel(const float* __restrict__ in, bf16_t* __restrict__ out, size_t n) {
    size_t i = (size_t)blockIdx.x * blockDim.x + threadIdx.x;
    size_t st = (size_t)gridDim.x * blockDim.x;
    for (; i < n; i += st) out[i] = (bf16_t)in[i];
}

// wkv_b (16,256,512) f32 -> nope part transposed [h][c][d<128] bf16,
//                          value part as-is   [h][d][c] bf16
__global__ void prep_wkvb_kernel(const float* __restrict__ w,
                                 bf16_t* __restrict__ nt, bf16_t* __restrict__ vv) {
    int id = blockIdx.x * blockDim.x + threadIdx.x;
    if (id >= H_ * 256 * KVR_) return;
    int h = id / (256 * KVR_);
    int rem = id % (256 * KVR_);
    int d = rem / KVR_, c = rem % KVR_;
    float val = w[id];
    if (d < DN_) nt[(size_t)h * (KVR_ * DN_) + c * DN_ + d] = (bf16_t)val;
    else         vv[(size_t)h * (DV_ * KVR_) + (d - DN_) * KVR_ + c] = (bf16_t)val;
}

// ---------------------------------------------------------------------------
// Generic batched bf16 GEMM: C[z][m][n] = sum_k A[m,k]*B[n,k] (+bias[n]).
// 128x128 block tile, 8 waves of 32x64. K stepped by 32. Both operand slabs
// (128x32) staged in LDS via double-buffered async-to-LDS DMA.
// ---------------------------------------------------------------------------
template <int OUT_BF16>
__global__ __launch_bounds__(256) void gemm_kernel(
    const bf16_t* __restrict__ A, const bf16_t* __restrict__ B,
    const float* __restrict__ bias, void* __restrict__ Cv,
    int M, int N, int K, int lda, int ldb, int ldc,
    long zsA, long zsB, long zsC)
{
    __shared__ alignas(16) bf16_t Asm[2][128 * 32];  // [buf][row][k] 8KB each
    __shared__ alignas(16) bf16_t Bsm[2][128 * 32];

    const int z = blockIdx.z;
    A += (size_t)z * zsA;
    B += (size_t)z * zsB;
    const int tid  = threadIdx.x;
    const int lane = tid & 31;
    const int w  = tid >> 5;
    const int mw = w & 3, nw = w >> 2;
    const int bm0 = blockIdx.y * 128;
    const int bn0 = blockIdx.x * 128;
    const int bm = bm0 + mw * 32;
    const int bn = bn0 + nw * 64;
    const int rl = lane & 15;
    const int ka = (lane >> 4) << 3;   // A lane K offset (0/8); == C high-row offset
    const int kb = (lane >> 4) << 4;   // unused (0/16); B fragments use ka

    // staging: 4096 elements per slab; round r: idx = r*2048 + tid*8
    unsigned gaoff[2], gboff[2], laoff[2], lboff[2];
    const unsigned asmB = lds_off32(&Asm[0][0]);
    const unsigned bsmB = lds_off32(&Bsm[0][0]);
#pragma unroll
    for (int r = 0; r < 2; ++r) {
        int idx = r * 2048 + tid * 8;
        int row = idx >> 5, col = idx & 31;
        int ra = bm0 + row; if (ra > M - 1) ra = M - 1;
        int rb = bn0 + row; if (rb > N - 1) rb = N - 1;
        gaoff[r] = (unsigned)(((size_t)ra * lda + col) * 2);
        gboff[r] = (unsigned)(((size_t)rb * ldb + col) * 2);
        laoff[r] = asmB + (unsigned)(r * 4096 + tid * 16);
        lboff[r] = bsmB + (unsigned)(r * 4096 + tid * 16);
    }
    // LDS fragment offsets (elements within one buf slab)
    int aoff[2], boff[4];
#pragma unroll
    for (int mi = 0; mi < 2; ++mi) aoff[mi] = (mw * 32 + mi * 16 + rl) * 32 + ka;
#pragma unroll
    for (int ni = 0; ni < 4; ++ni) boff[ni] = (nw * 64 + ni * 16 + rl) * 32 + ka;

    v8f acc[2][4] = {};

    // prologue: stage k=0 into buf 0   (ASYNCcnt += 4 per wave)
#pragma unroll
    for (int r = 0; r < 2; ++r) {
        async_g2l_16B(A, gaoff[r], laoff[r]);
        async_g2l_16B(B, gboff[r], lboff[r]);
    }

    int buf = 0;
    for (int k = 0; k < K; k += 32) {
        if (k + 32 < K) {
            unsigned kB = (unsigned)((k + 32) * 2);
            unsigned lbuf = (unsigned)((buf ^ 1) * 8192);
#pragma unroll
            for (int r = 0; r < 2; ++r) {
                async_g2l_16B(A, gaoff[r] + kB, laoff[r] + lbuf);
                async_g2l_16B(B, gboff[r] + kB, lboff[r] + lbuf);
            }
            wait_async_le4();   // previous stage landed; next still in flight
        } else {
            wait_async_le0();
        }
        __syncthreads();

        const bf16_t* Ab = &Asm[buf][0];
        const bf16_t* Bb = &Bsm[buf][0];
        v16bf af[2], bfr[4];
#pragma unroll
        for (int mi = 0; mi < 2; ++mi)
            af[mi] = make16(*(const bf16x8*)(Ab + aoff[mi]), *(const bf16x8*)(Ab + aoff[mi] + 16));
#pragma unroll
        for (int ni = 0; ni < 4; ++ni)
            bfr[ni] = make16(*(const bf16x8*)(Bb + boff[ni]), *(const bf16x8*)(Bb + boff[ni] + 16));
#pragma unroll
        for (int mi = 0; mi < 2; ++mi)
#pragma unroll
            for (int ni = 0; ni < 4; ++ni)
                acc[mi][ni] = wmma_bf16(af[mi], bfr[ni], acc[mi][ni]);
        __syncthreads();        // all waves done with `buf` before it is refilled
        buf ^= 1;
    }

    // C layout: VGPR r -> row m = r + (lane>=16 ? 8 : 0); col n = lane&15
#pragma unroll
    for (int ni = 0; ni < 4; ++ni) {
        int col = bn + ni * 16 + rl;
        if (col >= N) continue;
        float bv = bias ? bias[col] : 0.0f;
#pragma unroll
        for (int mi = 0; mi < 2; ++mi) {
#pragma unroll
            for (int r = 0; r < 8; ++r) {
                int row = bm + mi * 16 + r + ka;
                if (row >= M) continue;
                float v = acc[mi][ni][r] + bv;
                size_t idx = (size_t)z * zsC + (size_t)row * ldc + col;
                if (OUT_BF16) ((bf16_t*)Cv)[idx] = (bf16_t)v;
                else          ((float*)Cv)[idx] = v;
            }
        }
    }
}

// ---------------------------------------------------------------------------
// RMSNorm row kernel: out = w * x * rsqrt(mean(x^2)+1e-6), bf16 out.
// Optionally also writes transposed copy out_t[b][i][t] (for V^T of attention).
// ---------------------------------------------------------------------------
__global__ __launch_bounds__(256) void rms_kernel(
    const float* __restrict__ in, const float* __restrict__ w,
    bf16_t* __restrict__ out, bf16_t* __restrict__ out_t,
    int width, int istride, int ostride, int seq)
{
    __shared__ float red[256];
    int row = blockIdx.x;
    const float* x = in + (size_t)row * istride;
    float s = 0.f;
    for (int i = threadIdx.x; i < width; i += 256) { float v = x[i]; s += v * v; }
    red[threadIdx.x] = s;
    __syncthreads();
    for (int off = 128; off > 0; off >>= 1) {
        if (threadIdx.x < off) red[threadIdx.x] += red[threadIdx.x + off];
        __syncthreads();
    }
    float rs = rsqrtf(red[0] / (float)width + 1e-6f);
    for (int i = threadIdx.x; i < width; i += 256) {
        float v = x[i] * rs * w[i];
        out[(size_t)row * ostride + i] = (bf16_t)v;
        if (out_t) {
            int b = row / seq, t = row % seq;
            out_t[((size_t)b * width + i) * seq + t] = (bf16_t)v;
        }
    }
}

// ---------------------------------------------------------------------------
// RoPE helpers. inv_freq(i) = 10000^(-2i/64) = exp(-i * ln(10000)/32)
// ---------------------------------------------------------------------------
__device__ __forceinline__ float rope_inv(int i) {
    return __expf(-(float)i * 0.28782313662425572f);
}

// Split q (f32 [tok][3072]) -> q_nope bf16 [tok][h][128], roped q_pe into
// q_cat[tok][h][512..575]
__global__ void rope_q_kernel(const float* __restrict__ q,
                              bf16_t* __restrict__ q_nope, bf16_t* __restrict__ q_cat) {
    int token = blockIdx.x;
    int s = token & (S_ - 1);
    const float* qr = q + (size_t)token * (H_ * QKH);
    for (int j = threadIdx.x; j < H_ * QKH; j += blockDim.x) {
        int h = j / QKH, d = j % QKH;
        float v = qr[j];
        if (d < DN_) {
            q_nope[(size_t)token * (H_ * DN_) + h * DN_ + d] = (bf16_t)v;
        } else {
            int dp = d - DN_;
            float ang = (float)s * rope_inv(dp & 31);
            float rot = (dp < 32) ? -qr[h * QKH + DN_ + dp + 32]
                                  :  qr[h * QKH + DN_ + dp - 32];
            float o = v * __cosf(ang) + rot * __sinf(ang);
            q_cat[(size_t)token * (H_ * KCAT) + h * KCAT + KVR_ + dp] = (bf16_t)o;
        }
    }
}

// RoPE on k_pe (kv f32 cols 512..575) -> k_cat[tok][512..575]
__global__ void rope_k_kernel(const float* __restrict__ kv, bf16_t* __restrict__ k_cat) {
    int token = blockIdx.x;
    int s = token & (S_ - 1);
    int j = threadIdx.x; // 64 threads
    const float* kr = kv + (size_t)token * KCAT + KVR_;
    float v = kr[j];
    float ang = (float)s * rope_inv(j & 31);
    float rot = (j < 32) ? -kr[j + 32] : kr[j - 32];
    k_cat[(size_t)token * KCAT + KVR_ + j] = (bf16_t)(v * __cosf(ang) + rot * __sinf(ang));
}

// ---------------------------------------------------------------------------
// Flash attention: block = (b, h, 64-row s tile), 8 waves.
//   scores (K=576) via WMMA -> LDS -> online softmax -> P (bf16, A-layout in
//   LDS) -> O += P @ c_n^T via WMMA. O[64][512] split: wave w owns 64 cols.
// ---------------------------------------------------------------------------
__global__ __launch_bounds__(256) void mla_attn_kernel(
    const bf16_t* __restrict__ qcat,  // [tok][H][576]
    const bf16_t* __restrict__ kcat,  // [tok][576]
    const bf16_t* __restrict__ vt,    // [b][512][S]   (c_n transposed)
    const float*  __restrict__ mask,  // [b][S][S]
    bf16_t* __restrict__ oc)          // [tok][H][512]
{
    __shared__ alignas(16) float  S_lds[64][64];
    __shared__ alignas(16) bf16_t P_lds[64][64];
    __shared__ float  m_lds[64], l_lds[64], a_lds[64];

    const int lane = threadIdx.x & 31;
    const int w = threadIdx.x >> 5;
    const int b = blockIdx.z, h = blockIdx.y;
    const int s0 = blockIdx.x * 64;
    const int rl = lane & 15;
    const int ka = (lane >> 4) << 3;   // A-frag / C-row half offset
    const int kb = (lane >> 4) << 4;   // unused; B frags also use ka

    const bf16_t* qp = qcat + ((size_t)(b * S_ + s0) * H_ + h) * KCAT;  // row stride H_*KCAT
    const bf16_t* kp = kcat + (size_t)b * S_ * KCAT;
    const bf16_t* vp = vt + (size_t)b * KVR_ * S_;
    const float*  mp = mask + (size_t)b * S_ * S_ + (size_t)s0 * S_;
    bf16_t* op = oc + ((size_t)(b * S_ + s0) * H_ + h) * KVR_;

    if (threadIdx.x < 64) { m_lds[threadIdx.x] = -3.0e38f; l_lds[threadIdx.x] = 0.f; }
    __syncthreads();

    v8f O[4][4] = {};
    const int cw0 = w * 64;
    const float scale = 0.07216878364870323f;  // 1/sqrt(192)

    for (int t0 = 0; t0 <= s0; t0 += 64) {
        // ---- phase 1: 64x64 raw scores; wave w computes tiles 2w, 2w+1 ----
#pragma unroll
        for (int qi = 0; qi < 2; ++qi) {
            int tile = 2 * w + qi;
            int mi = tile >> 2, ti = tile & 3;
            const bf16_t* arow = qp + (size_t)(mi * 16 + rl) * (H_ * KCAT) + ka;
            const bf16_t* brow = kp + (size_t)(t0 + ti * 16 + rl) * KCAT + ka;
            v8f acc = {};
#pragma unroll
            for (int k = 0; k < KCAT; k += 32)
                acc = wmma_bf16(load_a_frag(arow, k), load_b_frag(brow, k), acc);
            int col = ti * 16 + rl;
#pragma unroll
            for (int r = 0; r < 8; ++r) {
                int row = mi * 16 + r + ka;
                S_lds[row][col] = acc[r] * scale + mp[(size_t)row * S_ + t0 + col];
            }
        }
        __syncthreads();

        // ---- phase 2: online softmax; wave owns rows w*8..w*8+7, 4 lanes/row
        {
            int row = w * 8 + (lane >> 2);
            int cb = (lane & 3) * 16;
            float vals[16];
            float mx = -3.0e38f;
#pragma unroll
            for (int i = 0; i < 16; ++i) { vals[i] = S_lds[row][cb + i]; mx = fmaxf(mx, vals[i]); }
#pragma unroll
            for (int off = 1; off < 4; off <<= 1) mx = fmaxf(mx, __shfl_xor(mx, off, 32));
            float mo = m_lds[row];
            float mn = fmaxf(mo, mx);
            float al = __expf(mo - mn);
            float sum = 0.f;
#pragma unroll
            for (int i = 0; i < 16; ++i) {
                float pv = __expf(vals[i] - mn);
                sum += pv;
                P_lds[row][cb + i] = (bf16_t)pv;
            }
#pragma unroll
            for (int off = 1; off < 4; off <<= 1) sum += __shfl_xor(sum, off, 32);
            if ((lane & 3) == 0) {
                l_lds[row] = l_lds[row] * al + sum;
                m_lds[row] = mn;
                a_lds[row] = al;
            }
        }
        __syncthreads();

        // ---- phase 3: rescale O, accumulate P @ V^T (K = 64) ----
#pragma unroll
        for (int mi = 0; mi < 4; ++mi) {
            float al[8];
#pragma unroll
            for (int r = 0; r < 8; ++r) al[r] = a_lds[mi * 16 + r + ka];
            const bf16_t* prow = &P_lds[mi * 16 + rl][ka];
            v16bf a0 = make16(*(const bf16x8*)(prow),      *(const bf16x8*)(prow + 16));
            v16bf a1 = make16(*(const bf16x8*)(prow + 32), *(const bf16x8*)(prow + 48));
#pragma unroll
            for (int ci = 0; ci < 4; ++ci) {
                v8f o = O[mi][ci];
#pragma unroll
                for (int r = 0; r < 8; ++r) o[r] *= al[r];
                const bf16_t* vrow = vp + (size_t)(cw0 + ci * 16 + rl) * S_ + t0 + ka;
                o = wmma_bf16(a0, make16(*(const bf16x8*)(vrow),      *(const bf16x8*)(vrow + 16)), o);
                o = wmma_bf16(a1, make16(*(const bf16x8*)(vrow + 32), *(const bf16x8*)(vrow + 48)), o);
                O[mi][ci] = o;
            }
        }
        __syncthreads();
    }

    // ---- final: normalize by 1/l, store bf16 ----
#pragma unroll
    for (int mi = 0; mi < 4; ++mi) {
        float linv[8];
#pragma unroll
        for (int r = 0; r < 8; ++r) {
            float l = l_lds[mi * 16 + r + ka];
            linv[r] = (l > 0.f) ? 1.f / l : 0.f;
        }
#pragma unroll
        for (int ci = 0; ci < 4; ++ci) {
            int col = cw0 + ci * 16 + rl;
#pragma unroll
            for (int r = 0; r < 8; ++r) {
                int row = mi * 16 + r + ka;
                op[(size_t)row * (H_ * KVR_) + col] = (bf16_t)(O[mi][ci][r] * linv[r]);
            }
        }
    }
}

// ---------------------------------------------------------------------------
extern "C" void kernel_launch(void* const* d_in, const int* in_sizes, int n_in,
                              void* d_out, int out_size, void* d_ws, size_t ws_size,
                              hipStream_t stream) {
    (void)in_sizes; (void)n_in; (void)out_size; (void)ws_size;
    const float* x        = (const float*)d_in[0];
    const float* mask     = (const float*)d_in[1];
    const float* wq_a_w   = (const float*)d_in[2];
    const float* wq_a_b   = (const float*)d_in[3];
    const float* q_norm_w = (const float*)d_in[4];
    const float* wq_b_w   = (const float*)d_in[5];
    const float* wq_b_b   = (const float*)d_in[6];
    const float* wkv_a_w  = (const float*)d_in[7];
    const float* wkv_a_b  = (const float*)d_in[8];
    const float* kv_norm_w= (const float*)d_in[9];
    const float* wkv_b_w  = (const float*)d_in[10];
    const float* wo_w     = (const float*)d_in[11];
    const float* wo_b     = (const float*)d_in[12];
    float* out = (float*)d_out;

    char* p = (char*)d_ws;
    auto alloc = [&](size_t n) { char* r = p; p += (n + 255) & ~(size_t)255; return r; };

    bf16_t* x_bf    = (bf16_t*)alloc((size_t)TOK * DIM * 2);
    bf16_t* wqa_bf  = (bf16_t*)alloc((size_t)QR_ * DIM * 2);
    bf16_t* wqb_bf  = (bf16_t*)alloc((size_t)H_ * QKH * QR_ * 2);
    bf16_t* wkva_bf = (bf16_t*)alloc((size_t)KCAT * DIM * 2);
    bf16_t* wo_bf   = (bf16_t*)alloc((size_t)DIM * DIM * 2);
    bf16_t* wkvb_nt = (bf16_t*)alloc((size_t)H_ * KVR_ * DN_ * 2);  // [h][c][d]
    bf16_t* wkvb_v  = (bf16_t*)alloc((size_t)H_ * DV_ * KVR_ * 2);  // [h][d][c]
    float*  q_a_f32 = (float*) alloc((size_t)TOK * QR_ * 4);
    bf16_t* q_a_n   = (bf16_t*)alloc((size_t)TOK * QR_ * 2);
    float*  q_f32   = (float*) alloc((size_t)TOK * H_ * QKH * 4);
    float*  kv_f32  = (float*) alloc((size_t)TOK * KCAT * 4);
    bf16_t* q_nope  = (bf16_t*)alloc((size_t)TOK * H_ * DN_ * 2);
    bf16_t* q_cat   = (bf16_t*)alloc((size_t)TOK * H_ * KCAT * 2); // [tok][h][576]
    bf16_t* k_cat   = (bf16_t*)alloc((size_t)TOK * KCAT * 2);      // [tok][576]
    bf16_t* c_n_t   = (bf16_t*)alloc((size_t)BS_ * KVR_ * S_ * 2); // [b][c][t]
    bf16_t* o_c     = (bf16_t*)alloc((size_t)TOK * H_ * KVR_ * 2); // [tok][h][512]
    bf16_t* o_head  = (bf16_t*)alloc((size_t)TOK * H_ * DV_ * 2);  // [tok][h*128+d]

    auto cvt = [&](const float* src, bf16_t* dst, size_t n) {
        cvt_kernel<<<dim3((unsigned)((n + 255) / 256)), dim3(256), 0, stream>>>(src, dst, n);
    };
    cvt(x,       x_bf,    (size_t)TOK * DIM);
    cvt(wq_a_w,  wqa_bf,  (size_t)QR_ * DIM);
    cvt(wq_b_w,  wqb_bf,  (size_t)H_ * QKH * QR_);
    cvt(wkv_a_w, wkva_bf, (size_t)KCAT * DIM);
    cvt(wo_w,    wo_bf,   (size_t)DIM * DIM);
    prep_wkvb_kernel<<<dim3((H_ * 256 * KVR_ + 255) / 256), dim3(256), 0, stream>>>(
        wkv_b_w, wkvb_nt, wkvb_v);

    // q_a = x @ wq_a^T + b  : [4096,1536] f32
    gemm_kernel<0><<<dim3(QR_ / 128, TOK / 128, 1), 256, 0, stream>>>(
        x_bf, wqa_bf, wq_a_b, q_a_f32, TOK, QR_, DIM, DIM, DIM, QR_, 0, 0, 0);
    // RMS over 1536 -> bf16
    rms_kernel<<<TOK, 256, 0, stream>>>(q_a_f32, q_norm_w, q_a_n, nullptr, QR_, QR_, QR_, S_);
    // q = q_a_n @ wq_b^T + b : [4096,3072] f32
    gemm_kernel<0><<<dim3(H_ * QKH / 128, TOK / 128, 1), 256, 0, stream>>>(
        q_a_n, wqb_bf, wq_b_b, q_f32, TOK, H_ * QKH, QR_, QR_, QR_, H_ * QKH, 0, 0, 0);
    // kv = x @ wkv_a^T + b : [4096,576] f32
    gemm_kernel<0><<<dim3((KCAT + 127) / 128, TOK / 128, 1), 256, 0, stream>>>(
        x_bf, wkva_bf, wkv_a_b, kv_f32, TOK, KCAT, DIM, DIM, DIM, KCAT, 0, 0, 0);

    // split q -> q_nope (bf16) + roped q_pe into q_cat[...,512:576]
    rope_q_kernel<<<TOK, 256, 0, stream>>>(q_f32, q_nope, q_cat);
    // c_n = RMS(c) -> k_cat[...,0:512] and transposed c_n_t
    rms_kernel<<<TOK, 256, 0, stream>>>(kv_f32, kv_norm_w, k_cat, c_n_t, KVR_, KCAT, KCAT, S_);
    // roped k_pe -> k_cat[...,512:576]
    rope_k_kernel<<<TOK, 64, 0, stream>>>(kv_f32, k_cat);

    // q_abs[tok][h][0:512] = q_nope[tok][h] @ wkv_b_nope[h]  (per-head, batched)
    gemm_kernel<1><<<dim3(KVR_ / 128, TOK / 128, H_), 256, 0, stream>>>(
        q_nope, wkvb_nt, nullptr, q_cat,
        TOK, KVR_, DN_, H_ * DN_, DN_, H_ * KCAT,
        (long)DN_, (long)(KVR_ * DN_), (long)KCAT);

    // flash attention -> o_c
    mla_attn_kernel<<<dim3(S_ / 64, H_, BS_), 256, 0, stream>>>(
        q_cat, k_cat, c_n_t, mask, o_c);

    // per-head V projection: o_head[tok][h*128+d] = o_c[tok][h] @ wkv_b_v[h]^T
    gemm_kernel<1><<<dim3(1, TOK / 128, H_), 256, 0, stream>>>(
        o_c, wkvb_v, nullptr, o_head,
        TOK, DV_, KVR_, H_ * KVR_, KVR_, H_ * DV_,
        (long)KVR_, (long)(DV_ * KVR_), (long)DV_);

    // out = o_head @ wo^T + b : f32 -> d_out
    gemm_kernel<0><<<dim3(DIM / 128, TOK / 128, 1), 256, 0, stream>>>(
        o_head, wo_bf, wo_b, out, TOK, DIM, DIM, DIM, DIM, DIM, 0, 0, 0);
}
